// Model_27324581937589
// MI455X (gfx1250) — matrix operand order repaired and checked by reading, not compiled, not run on hardware
//
#include <hip/hip_runtime.h>
#include <hip/hip_bf16.h>

// ---------------------------------------------------------------------------
// Types for CDNA5 WMMA (wave32): D(16x16,f32) = A(16x32,bf16) x B(32x16,bf16) + C
// ---------------------------------------------------------------------------
typedef __attribute__((ext_vector_type(16))) __bf16 v16bf;
typedef __attribute__((ext_vector_type(8)))  float  v8f;

#define DI_N 2048
#define DR_N 1024
#define P_N  4096
#define G_N  4096
#define K1   128
#define K2   64
#define K3   32
#define O_DI1 2048
#define O_DI3 4096
#define O_DR2 6144
#define O_DR1 7168
#define O_DR3 8192
#define O_P3  9216
#define O_P2  13312
#define O_G3  17408
#define O_G2  21504
#define N_RGCN 25600
#define MT 4   // M subtiles (16 rows each) per block => 64-row block tile

// Pack two fp32 -> two bf16 (round-to-nearest-even) in one dword.
__device__ __forceinline__ unsigned pack_bf16(float lo, float hi) {
  unsigned ua = __float_as_uint(lo);
  unsigned ub = __float_as_uint(hi);
  ua += 0x7FFFu + ((ua >> 16) & 1u);
  ub += 0x7FFFu + ((ub >> 16) & 1u);
  return (ua >> 16) | (ub & 0xFFFF0000u);
}

union frag_u {
  unsigned u[8];
  v16bf    v;
};

// ---------------------------------------------------------------------------
// GEMM: C[M,N] = act(A[M,K] @ W[K,N] + bias), fp32 in/out, bf16 WMMA compute.
// Block tile = 64 x (nw*16). LDS holds fragments pre-packed in WMMA register
// order (two ds_load_b128 per fragment). Each wave keeps one B fragment in
// registers and issues MT=4 WMMAs per K-step (one per 16-row A subtile),
// amortizing staging + barriers over 4 matrix ops.
// M%64==0, K%32==0, N in {32,64,128} for all call sites.
// ---------------------------------------------------------------------------
__global__ void gemm_bf16_wmma(const float* __restrict__ A,
                               const float* __restrict__ W,
                               const float* __restrict__ bias,
                               float* __restrict__ C,
                               int M, int N, int K, int relu) {
  __shared__ __align__(32) unsigned sAq[MT * 256];     // 4 KB: A fragments (4 subtiles)
  __shared__ __align__(32) unsigned sBq[8 * 256];      // 8 KB: per-wave B fragments
  const int nw    = blockDim.x >> 5;    // waves per block
  const int ncols = nw << 4;            // N columns handled per block
  const int wid   = threadIdx.x >> 5;
  const int lane  = threadIdx.x & 31;
  const int rowBase = blockIdx.x * (16 * MT);
  const int colBase = blockIdx.y * ncols;

  v8f acc[MT] = {};
  for (int k0 = 0; k0 < K; k0 += 32) {
    // --- Stage A fragments: MT*256 packed dwords, fragment-ordered.
    // Subtile t, lane L, dword j holds K pair: j<4 -> khalf+2j ; j>=4 -> 16+khalf+2(j-4).
    for (int i = threadIdx.x; i < MT * 256; i += blockDim.x) {
      int t = i >> 8, rem = i & 255;
      int L = rem >> 3, j = rem & 7;
      int m = t * 16 + (L & 15);
      int khalf = (L < 16) ? 0 : 8;
      int k = (j < 4) ? (khalf + 2 * j) : (16 + khalf + 2 * (j - 4));
      const float2 f = *(const float2*)&A[(size_t)(rowBase + m) * K + (k0 + k)];
      sAq[i] = pack_bf16(f.x, f.y);
    }
    // --- Stage B fragments: nw*256 packed dwords.
    // Lane L, dword j holds K pair: j<4 -> kg+2j ; j>=4 -> 16+kg+2(j-4); kg = (L<16)?0:8.
    for (int i = threadIdx.x; i < (nw << 8); i += blockDim.x) {
      int w = i >> 8, rem = i & 255;
      int L = rem >> 3, j = rem & 7;
      int n = colBase + w * 16 + (L & 15);
      int kg = (L < 16) ? 0 : 8;
      int k = k0 + ((j < 4) ? (kg + 2 * j) : (16 + kg + 2 * (j - 4)));
      sBq[i] = pack_bf16(W[(size_t)k * N + n], W[(size_t)(k + 1) * N + n]);
    }
    // Prefetch next A tile rows (global_prefetch path)
    if (k0 + 32 < K && threadIdx.x < 64)
      __builtin_prefetch(&A[(size_t)(rowBase + threadIdx.x) * K + k0 + 32], 0, 1);
    __syncthreads();

    v16bf bf = ((const v16bf*)sBq)[wid * 32 + lane];   // 2x ds_load_b128, reused 4x
#pragma unroll
    for (int t = 0; t < MT; ++t) {
      v16bf af = ((const v16bf*)sAq)[t * 32 + lane];   // 2x ds_load_b128
      acc[t] = __builtin_amdgcn_wmma_f32_16x16x32_bf16(false, af, false, bf,
                                                       (short)0, acc[t], false, false);
    }
    __syncthreads();
  }

  // C layout: VGPR r, lanes 0-15 -> M=r, N=lane; lanes 16-31 -> M=8+r
  const int ccol  = colBase + wid * 16 + (lane & 15);
  const int rbase = (lane < 16) ? 0 : 8;
  float bv = bias ? bias[ccol] : 0.0f;
#pragma unroll
  for (int t = 0; t < MT; ++t) {
#pragma unroll
    for (int r = 0; r < 8; ++r) {
      float v = acc[t][r] + bv;
      if (relu) v = fmaxf(v, 0.0f);
      C[(size_t)(rowBase + t * 16 + rbase + r) * N + ccol] = v;
    }
  }
}

// ---------------------------------------------------------------------------
// C[M,N] = A[M,32] @ B[N,32]^T  (K=32 => single bf16 WMMA per 16x16 tile).
// Fragments built from four float4 global loads per matrix, packed in regs.
// Block = 8 waves covering a 16x128 C tile.
// ---------------------------------------------------------------------------
__global__ void gemm_abt_wmma(const float* __restrict__ A,
                              const float* __restrict__ B,
                              float* __restrict__ C, int M, int N) {
  const int wid  = threadIdx.x >> 5;
  const int lane = threadIdx.x & 31;
  const int rowBase = blockIdx.x * 16;
  const int colBase = blockIdx.y * 128 + wid * 16;

  const int m     = rowBase + (lane & 15);
  const int khalf = (lane < 16) ? 0 : 8;
  frag_u Ua, Ub;
  {
    const float* ap = A + (size_t)m * 32;
    float4 a0 = *(const float4*)(ap + khalf);
    float4 a1 = *(const float4*)(ap + khalf + 4);
    float4 a2 = *(const float4*)(ap + 16 + khalf);
    float4 a3 = *(const float4*)(ap + 16 + khalf + 4);
    Ua.u[0] = pack_bf16(a0.x, a0.y); Ua.u[1] = pack_bf16(a0.z, a0.w);
    Ua.u[2] = pack_bf16(a1.x, a1.y); Ua.u[3] = pack_bf16(a1.z, a1.w);
    Ua.u[4] = pack_bf16(a2.x, a2.y); Ua.u[5] = pack_bf16(a2.z, a2.w);
    Ua.u[6] = pack_bf16(a3.x, a3.y); Ua.u[7] = pack_bf16(a3.z, a3.w);
  }
  const int n  = colBase + (lane & 15);
  const int kg = (lane < 16) ? 0 : 8;
  {
    const float* bp = B + (size_t)n * 32 + kg;
    float4 b0 = *(const float4*)(bp);
    float4 b1 = *(const float4*)(bp + 4);
    float4 b2 = *(const float4*)(bp + 16);
    float4 b3 = *(const float4*)(bp + 20);
    Ub.u[0] = pack_bf16(b0.x, b0.y); Ub.u[1] = pack_bf16(b0.z, b0.w);
    Ub.u[2] = pack_bf16(b1.x, b1.y); Ub.u[3] = pack_bf16(b1.z, b1.w);
    Ub.u[4] = pack_bf16(b2.x, b2.y); Ub.u[5] = pack_bf16(b2.z, b2.w);
    Ub.u[6] = pack_bf16(b3.x, b3.y); Ub.u[7] = pack_bf16(b3.z, b3.w);
  }
  v8f acc = {};
  acc = __builtin_amdgcn_wmma_f32_16x16x32_bf16(false, Ua.v, false, Ub.v,
                                                (short)0, acc, false, false);
  const int rbase = (lane < 16) ? 0 : 8;
#pragma unroll
  for (int r = 0; r < 8; ++r)
    C[(size_t)(rowBase + rbase + r) * N + n] = acc[r];
}

// ---------------------------------------------------------------------------
// Graph scatter / normalization kernels
// ---------------------------------------------------------------------------
__global__ void deg_count(const int* __restrict__ e, int E,
                          float* __restrict__ deg, int bidir) {
  int i = blockIdx.x * blockDim.x + threadIdx.x;
  if (i >= E) return;
  atomicAdd(&deg[e[E + i]], 1.0f);
  if (bidir) atomicAdd(&deg[e[i]], 1.0f);
}

// deg -> rsqrt(max(deg + 1 (self loop), 1))
__global__ void dinv_kernel(float* __restrict__ deg, int n) {
  int i = blockIdx.x * blockDim.x + threadIdx.x;
  if (i >= n) return;
  deg[i] = rsqrtf(fmaxf(deg[i] + 1.0f, 1.0f));
}

// acc[dst] += h[src] * dinv[src]*dinv[dst]  (and reverse if bidir)
__global__ void gcn_scatter(const int* __restrict__ e, int E,
                            const float* __restrict__ hA,
                            const float* __restrict__ hB, int split,
                            const float* __restrict__ dinv,
                            float* __restrict__ acc, int bidir) {
  int edge = blockIdx.x * (blockDim.x >> 5) + (threadIdx.x >> 5);
  int lane = threadIdx.x & 31;
  if (edge >= E) return;
  int s = e[edge], d = e[E + edge];
  float nrm = dinv[s] * dinv[d];
  const float* hs = (s < split) ? hA + (size_t)s * K1 : hB + (size_t)(s - split) * K1;
  float* ad = acc + (size_t)d * K1;
#pragma unroll
  for (int c = lane; c < K1; c += 32) atomicAdd(&ad[c], hs[c] * nrm);
  if (bidir) {
    const float* hd = (d < split) ? hA + (size_t)d * K1 : hB + (size_t)(d - split) * K1;
    float* as = acc + (size_t)s * K1;
#pragma unroll
    for (int c = lane; c < K1; c += 32) atomicAdd(&as[c], hd[c] * nrm);
  }
}

// out = relu(acc + h*dinv^2 + bias)   (self-loop term + bias + activation)
__global__ void gcn_finalize(const float* __restrict__ acc,
                             const float* __restrict__ h,
                             const float* __restrict__ dinv,
                             const float* __restrict__ bias,
                             float* __restrict__ out, int n) {
  int i = blockIdx.x * blockDim.x + threadIdx.x;
  if (i >= n * K1) return;
  int r = i >> 7, c = i & 127;
  float dv = dinv[r];
  out[i] = fmaxf(acc[i] + h[i] * dv * dv + bias[c], 0.0f);
}

__global__ void rel_count(const int* __restrict__ e, int E,
                          float* __restrict__ cnt, int rev) {
  int i = blockIdx.x * blockDim.x + threadIdx.x;
  if (i >= E) return;
  int d = rev ? e[i] : e[E + i];
  atomicAdd(&cnt[d], 1.0f);
}

// out[dst] += msg[src] / max(cnt[dst],1)
__global__ void rel_scatter(const int* __restrict__ e, int E,
                            const float* __restrict__ msg,
                            const float* __restrict__ cnt,
                            float* __restrict__ out, int rev) {
  int edge = blockIdx.x * (blockDim.x >> 5) + (threadIdx.x >> 5);
  int lane = threadIdx.x & 31;
  if (edge >= E) return;
  int s = rev ? e[E + edge] : e[edge];
  int d = rev ? e[edge] : e[E + edge];
  float inv = 1.0f / fmaxf(cnt[d], 1.0f);
  const float* ms = msg + (size_t)s * K1;
  float* od = out + (size_t)d * K1;
#pragma unroll
  for (int c = lane; c < K1; c += 32) atomicAdd(&od[c], ms[c] * inv);
}

__global__ void relu_inplace(float* __restrict__ x, int n) {
  int i = blockIdx.x * blockDim.x + threadIdx.x;
  if (i < n) x[i] = fmaxf(x[i], 0.0f);
}

__global__ void sum3(const float* a, const float* b, const float* c,
                     float* o, int n) {
  int i = blockIdx.x * blockDim.x + threadIdx.x;
  if (i < n) o[i] = a[i] + b[i] + c[i];
}

__global__ void sum2(const float* a, const float* b, float* o, int n) {
  int i = blockIdx.x * blockDim.x + threadIdx.x;
  if (i < n) o[i] = a[i] + b[i];
}

// ---------------------------------------------------------------------------
// Host orchestration
// ---------------------------------------------------------------------------
extern "C" void kernel_launch(void* const* d_in, const int* in_sizes, int n_in,
                              void* d_out, int out_size, void* d_ws, size_t ws_size,
                              hipStream_t stream) {
  (void)n_in; (void)out_size; (void)ws_size;
  auto F = [&](int i) { return (const float*)d_in[i]; };
  auto I = [&](int i) { return (const int*)d_in[i]; };

  // params flattened jax-tree style (sorted keys) after the 20 positional inputs
  const int P_GCN_DIDI_W = 20, P_GCN_DIDI_B = 21;
  const int P_GCN_DIG_W  = 22, P_GCN_DIG_B  = 23;
  const int P_GCN_DIP_W  = 24, P_GCN_DIP_B  = 25;
  const int P_GCN_DRDI_W = 26, P_GCN_DRDI_B = 27;
  const int P_GCN_DRDR_W = 28, P_GCN_DRDR_B = 29;
  const int P_GCN_DRG_W  = 30, P_GCN_DRG_B  = 31;
  const int P_GCN_DRP_W  = 32, P_GCN_DRP_B  = 33;
  const int P_LIN_DI_W = 34, P_LIN_DI_B = 35;
  const int P_LIN_DR_W = 36, P_LIN_DR_B = 37;
  const int P_LIN_G_W  = 38, P_LIN_G_B  = 39;
  const int P_LIN_P_W  = 40, P_LIN_P_B  = 41;
  const int P_MLP_DI = 42, P_MLP_DR = 48, P_MLP_G = 54, P_MLP_P = 60;
  const int P_WREL = 66, P_WROOT = 67, P_RB = 68;

  // Workspace carve-up
  float* ws = (float*)d_ws;
  size_t off = 0;
  auto A_ = [&](size_t n) { float* p = ws + off; off += n; return p; };
  float* X    = A_((size_t)N_RGCN * K1);
  float* MSG  = A_((size_t)N_RGCN * K1);
  float* ACC  = A_((size_t)N_RGCN * K1);
  float* HA   = A_((size_t)4096 * K1);
  float* HB   = A_((size_t)4096 * K1);
  float* P1   = A_((size_t)P_N * K1);
  float* G1   = A_((size_t)G_N * K1);
  float* DEG  = A_(N_RGCN);
  float* CNT  = A_(N_RGCN);
  float* DSUM = A_((size_t)DI_N * K1);
  float* RSUM = A_((size_t)DR_N * K1);
  float* PSUM = A_((size_t)P_N * K1);
  float* GSUM = A_((size_t)G_N * K1);
  float* M1B  = A_((size_t)4096 * K1);
  float* M2B  = A_((size_t)4096 * K2);
  float* MDI  = A_((size_t)DI_N * K3);
  float* MDR  = A_((size_t)DR_N * K3);
  float* MPm  = A_((size_t)P_N * K3);
  float* MGm  = A_((size_t)G_N * K3);

  auto launch_gemm = [&](const float* Am, int M, int K, const float* Wt, int N,
                         const float* bias, int relu, float* Cm) {
    int nw = (N >= 128) ? 8 : (N / 16);
    dim3 grid(M / (16 * MT), N / (nw * 16));
    gemm_bf16_wmma<<<grid, dim3(nw * 32), 0, stream>>>(Am, Wt, bias, Cm, M, N, K, relu);
  };

  // ---- dense input projections (relu(x@W+b)) ----
  launch_gemm(F(0), DI_N, DI_N, F(P_LIN_DI_W), K1, F(P_LIN_DI_B), 1, X);                       // di2
  launch_gemm(F(1), DR_N, DR_N, F(P_LIN_DR_W), K1, F(P_LIN_DR_B), 1, X + (size_t)O_DR2 * K1); // dr2
  launch_gemm(F(2), P_N, P_N, F(P_LIN_P_W), K1, F(P_LIN_P_B), 1, P1);                          // p1
  launch_gemm(F(3), G_N, G_N, F(P_LIN_G_W), K1, F(P_LIN_G_B), 1, G1);                          // g1

  // ---- homogeneous GCNs ----
  auto gcn_single = [&](const float* data, int n, int dataK, const int* e, int E,
                        const float* Wt, const float* b, float* outSlice) {
    launch_gemm(data, n, dataK, Wt, K1, nullptr, 0, HA);
    hipMemsetAsync(DEG, 0, (size_t)n * sizeof(float), stream);
    deg_count<<<dim3((E + 255) / 256), 256, 0, stream>>>(e, E, DEG, 0);
    dinv_kernel<<<dim3((n + 255) / 256), 256, 0, stream>>>(DEG, n);
    hipMemsetAsync(ACC, 0, (size_t)n * K1 * sizeof(float), stream);
    gcn_scatter<<<dim3((E + 7) / 8), 256, 0, stream>>>(e, E, HA, HA, n, DEG, ACC, 0);
    gcn_finalize<<<dim3((n * K1 + 255) / 256), 256, 0, stream>>>(ACC, HA, DEG, b, outSlice, n);
  };
  gcn_single(F(0), DI_N, DI_N, I(4), in_sizes[4] / 2, F(P_GCN_DIDI_W), F(P_GCN_DIDI_B),
             X + (size_t)O_DI1 * K1);                                                           // di1
  gcn_single(F(1), DR_N, DR_N, I(5), in_sizes[5] / 2, F(P_GCN_DRDR_W), F(P_GCN_DRDR_B),
             X + (size_t)O_DR1 * K1);                                                           // dr1

  // ---- bipartite GCNs (bidirectional edges, concat handled via split) ----
  auto gcn_bip = [&](const float* Asrc, int nA, const float* Bsrc, int nB,
                     const int* e, int E, const float* Wt, const float* b,
                     float* dstA, float* dstB) {
    launch_gemm(Asrc, nA, K1, Wt, K1, nullptr, 0, HA);
    launch_gemm(Bsrc, nB, K1, Wt, K1, nullptr, 0, HB);
    int n = nA + nB;
    hipMemsetAsync(DEG, 0, (size_t)n * sizeof(float), stream);
    deg_count<<<dim3((E + 255) / 256), 256, 0, stream>>>(e, E, DEG, 1);
    dinv_kernel<<<dim3((n + 255) / 256), 256, 0, stream>>>(DEG, n);
    hipMemsetAsync(ACC, 0, (size_t)n * K1 * sizeof(float), stream);
    gcn_scatter<<<dim3((E + 7) / 8), 256, 0, stream>>>(e, E, HA, HB, nA, DEG, ACC, 1);
    if (dstA)
      gcn_finalize<<<dim3((nA * K1 + 255) / 256), 256, 0, stream>>>(ACC, HA, DEG, b, dstA, nA);
    if (dstB)
      gcn_finalize<<<dim3((nB * K1 + 255) / 256), 256, 0, stream>>>(
          ACC + (size_t)nA * K1, HB, DEG + nA, b, dstB, nB);
  };
  const float* di2 = X;
  const float* dr2 = X + (size_t)O_DR2 * K1;
  gcn_bip(di2, DI_N, dr2, DR_N, I(6), in_sizes[6] / 2, F(P_GCN_DRDI_W), F(P_GCN_DRDI_B),
          X + (size_t)O_DI3 * K1, X + (size_t)O_DR3 * K1);                                      // di3,dr3
  gcn_bip(di2, DI_N, P1, P_N, I(7), in_sizes[7] / 2, F(P_GCN_DIP_W), F(P_GCN_DIP_B),
          nullptr, X + (size_t)O_P3 * K1);                                                      // p3
  gcn_bip(dr2, DR_N, P1, P_N, I(8), in_sizes[8] / 2, F(P_GCN_DRP_W), F(P_GCN_DRP_B),
          nullptr, X + (size_t)O_P2 * K1);                                                      // p2
  gcn_bip(di2, DI_N, G1, G_N, I(9), in_sizes[9] / 2, F(P_GCN_DIG_W), F(P_GCN_DIG_B),
          nullptr, X + (size_t)O_G3 * K1);                                                      // g3
  gcn_bip(dr2, DR_N, G1, G_N, I(10), in_sizes[10] / 2, F(P_GCN_DRG_W), F(P_GCN_DRG_B),
          nullptr, X + (size_t)O_G2 * K1);                                                      // g2

  // ---- RGCN: out = x@Wroot + b + sum_r segmean_r(x@Wrel[r]) ----
  launch_gemm(X, N_RGCN, K1, F(P_WROOT), K1, F(P_RB), 0, ACC);
  const float* Wrel = F(P_WREL);
  const int relIn[9] = {11, 12, 13, 16, 14, 15, 17, 18, 19}; // forward-pass relation order
  for (int r = 0; r < 18; ++r) {
    int idx = relIn[r >> 1];
    int rev = r & 1;
    const int* e = I(idx);
    int E = in_sizes[idx] / 2;
    launch_gemm(X, N_RGCN, K1, Wrel + (size_t)r * K1 * K1, K1, nullptr, 0, MSG);
    hipMemsetAsync(CNT, 0, (size_t)N_RGCN * sizeof(float), stream);
    rel_count<<<dim3((E + 255) / 256), 256, 0, stream>>>(e, E, CNT, rev);
    rel_scatter<<<dim3((E + 7) / 8), 256, 0, stream>>>(e, E, MSG, CNT, ACC, rev);
  }
  relu_inplace<<<dim3((N_RGCN * K1 + 255) / 256), 256, 0, stream>>>(ACC, N_RGCN * K1);

  // ---- per-type block sums ----
  sum3<<<dim3((DI_N * K1 + 255) / 256), 256, 0, stream>>>(
      ACC, ACC + (size_t)O_DI1 * K1, ACC + (size_t)O_DI3 * K1, DSUM, DI_N * K1);
  sum3<<<dim3((DR_N * K1 + 255) / 256), 256, 0, stream>>>(
      ACC + (size_t)O_DR2 * K1, ACC + (size_t)O_DR1 * K1, ACC + (size_t)O_DR3 * K1,
      RSUM, DR_N * K1);
  sum2<<<dim3((P_N * K1 + 255) / 256), 256, 0, stream>>>(
      ACC + (size_t)O_P3 * K1, ACC + (size_t)O_P2 * K1, PSUM, P_N * K1);
  sum2<<<dim3((G_N * K1 + 255) / 256), 256, 0, stream>>>(
      ACC + (size_t)O_G3 * K1, ACC + (size_t)O_G2 * K1, GSUM, G_N * K1);

  // ---- MLPs: 128 -> 128 -> 64 -> 32, relu each layer ----
  auto mlp = [&](const float* in, int M, int pb, float* outp) {
    launch_gemm(in, M, K1, F(pb + 0), K1, F(pb + 1), 1, M1B);
    launch_gemm(M1B, M, K1, F(pb + 2), K2, F(pb + 3), 1, M2B);
    launch_gemm(M2B, M, K2, F(pb + 4), K3, F(pb + 5), 1, outp);
  };
  mlp(DSUM, DI_N, P_MLP_DI, MDI);
  mlp(RSUM, DR_N, P_MLP_DR, MDR);
  mlp(PSUM, P_N, P_MLP_P, MPm);
  mlp(GSUM, G_N, P_MLP_G, MGm);

  // ---- outputs: (mdr@mdi.T, mp@mdi.T, mp@mdr.T, mg@mdi.T, mg@mdr.T) ----
  float* out = (float*)d_out;
  float* o1 = out;
  float* o2 = o1 + (size_t)DR_N * DI_N;
  float* o3 = o2 + (size_t)P_N * DI_N;
  float* o4 = o3 + (size_t)P_N * DR_N;
  float* o5 = o4 + (size_t)G_N * DI_N;
  auto abt = [&](const float* Am, int M, const float* Bm, int N, float* Cm) {
    dim3 grid(M / 16, N / 128);
    gemm_abt_wmma<<<grid, dim3(256), 0, stream>>>(Am, Bm, Cm, M, N);
  };
  abt(MDR, DR_N, MDI, DI_N, o1);
  abt(MPm, P_N, MDI, DI_N, o2);
  abt(MPm, P_N, MDR, DR_N, o3);
  abt(MGm, G_N, MDI, DI_N, o4);
  abt(MGm, G_N, MDR, DR_N, o5);
}
